// _ScaledDotProductAttention_20263655703291
// MI455X (gfx1250) — matrix operand order repaired and checked by reading, not compiled, not run on hardware
//
#include <hip/hip_runtime.h>
#include <hip/hip_bf16.h>

// Scaled dot-product attention for MI455X (gfx1250, wave32, WMMA).
// Outputs (out, attn) concatenated flat in d_out. Memory-roofline bound by the
// 512MB attn store (~28us at 23.3 TB/s); matmuls run on v_wmma_f32_16x16x32_f16
// with f32 accumulation so compute stays far under the HBM floor. The attn
// stream-out uses non-temporal b128 stores so it doesn't evict the K/V working
// set (256KB per (b,h), reused by 64 workgroups) from the 192MB L2.

typedef __attribute__((ext_vector_type(16))) _Float16 v16h;
typedef __attribute__((ext_vector_type(8)))  float    v8f;
typedef __attribute__((ext_vector_type(4)))  float    v4f;   // native vector for NT stores

#define B_  8
#define H_  16
#define S_  1024
#define D_  64
#define QTILE 16
#define WAVES 4
#define NTHREADS 128   // 4 wave32s

__global__ __launch_bounds__(NTHREADS)
void attn_fused_kernel(const float* __restrict__ Q,
                       const float* __restrict__ K,
                       const float* __restrict__ V,
                       float* __restrict__ out,
                       float* __restrict__ attn)
{
    // Unnormalized exp-scores for this 16-row q tile (f16): 32 KB.
    __shared__ _Float16 sE[QTILE * S_];
    __shared__ float    sRed[NTHREADS];
    __shared__ float    sInv[QTILE];

    const int tid  = threadIdx.x;
    const int wave = tid >> 5;
    const int lane = tid & 31;
    const int half = lane >> 4;   // wave32 half: selects K/M sub-range per ISA layouts
    const int m    = lane & 15;

    const int bh    = blockIdx.x / (S_ / QTILE);
    const int qbase = (blockIdx.x % (S_ / QTILE)) * QTILE;

    const float scale = 0.125f;   // 1/sqrt(D), D = 64

    const float* Qb = Q + ((size_t)bh * S_ + qbase) * D_;
    const float* Kb = K + (size_t)bh * S_ * D_;
    const float* Vb = V + (size_t)bh * S_ * D_;

    // ---------------- Phase 1: sE = exp(scale * Q K^T) ----------------
    // A fragments (Q tile, 16x32 f16), this code's contiguous k order (B matches):
    //   lanes 0-15: M=lane, k = {0..15}+kc ; lanes 16-31: k = {16..31}+kc
    v16h aq0, aq1;
    {
        const float* qrow = Qb + (size_t)m * D_;
        #pragma unroll
        for (int j = 0; j < 8; ++j) {
            aq0[j]     = (_Float16)qrow[ 0 + 16*half + j];
            aq0[j + 8] = (_Float16)qrow[ 8 + 16*half + j];
            aq1[j]     = (_Float16)qrow[32 + 16*half + j];
            aq1[j + 8] = (_Float16)qrow[40 + 16*half + j];
        }
    }

    for (int t = wave; t < S_ / 16; t += WAVES) {
        const int kbase = t * 16;
        const float* krow = Kb + (size_t)(kbase + m) * D_;
        if (t + WAVES < S_ / 16)  // prefetch next K tile row -> global_prefetch_b8
            __builtin_prefetch(Kb + (size_t)(kbase + WAVES * 16 + m) * D_, 0, 0);

        // B fragments (K^T chunk, 32x16 f16): lanes 0-15 hold k-rows 0..15
        // (column n = lane), lanes 16-31 hold k-rows 16..31. Element B[d,n] =
        // K[kbase+n, kc+d] -> each lane reads 16 contiguous f32 from its K row.
        v16h b0, b1;
        #pragma unroll
        for (int j = 0; j < 16; ++j) {
            b0[j] = (_Float16)krow[ 0 + 16*half + j];
            b1[j] = (_Float16)krow[32 + 16*half + j];
        }

        v8f c = {};
        c = __builtin_amdgcn_wmma_f32_16x16x32_f16(false, aq0, false, b0, (short)0, c, false, false);
        c = __builtin_amdgcn_wmma_f32_16x16x32_f16(false, aq1, false, b1, (short)0, c, false, false);

        // C layout: VGPR r, lanes 0-15 -> (M=r, N=lane); lanes 16-31 -> (M=r+8).
        #pragma unroll
        for (int r = 0; r < 8; ++r) {
            float e = __expf(c[r] * scale);
            sE[(r + 8 * half) * S_ + kbase + m] = (_Float16)e;
        }
    }
    __syncthreads();

    // ---------------- Row sums -> 1/sum ----------------
    {
        const int row  = tid >> 3;          // 8 threads per row
        const int part = tid & 7;
        const _Float16* p = &sE[row * S_ + part * (S_ / 8)];
        float s = 0.f;
        for (int i = 0; i < S_ / 8; ++i) s += (float)p[i];
        sRed[tid] = s;
    }
    __syncthreads();
    if (tid < QTILE) {
        float s = 0.f;
        #pragma unroll
        for (int j = 0; j < 8; ++j) s += sRed[tid * 8 + j];
        sInv[tid] = 1.0f / s;
    }
    __syncthreads();

    // ------- Normalized attn -> global: wide non-temporal stream-out -------
    // 16 contiguous rows => tile is one flat 16384-float block; b128 NT stores.
    {
        float* attnBase = attn + ((size_t)bh * S_ + qbase) * S_;
        #pragma unroll 4
        for (int i = 0; i < (QTILE * S_) / (NTHREADS * 4); ++i) {
            const int idx4 = (i * NTHREADS + tid) * 4;
            const float inv = sInv[idx4 >> 10];
            v4f v;
            v.x = (float)sE[idx4 + 0] * inv;
            v.y = (float)sE[idx4 + 1] * inv;
            v.z = (float)sE[idx4 + 2] * inv;
            v.w = (float)sE[idx4 + 3] * inv;
            __builtin_nontemporal_store(v, (v4f*)(attnBase + idx4));
        }
    }

    // ---------------- Phase 2: out = (P V) * (1/rowsum) ----------------
    // Each wave owns 16 columns of D. A comes from LDS probs (f16), B from V.
    // Softmax normalization is a uniform row scale, so we fold 1/rowsum into
    // the final C fragment instead of rewriting LDS.
    {
        const int nb = wave * 16;
        v8f c = {};
        for (int kc = 0; kc < S_; kc += 32) {
            v16h a, b;
            const _Float16* prow = &sE[m * S_ + kc];
            #pragma unroll
            for (int j = 0; j < 8; ++j) {
                a[j]     = prow[ 0 + 16 * half + j];
                a[j + 8] = prow[ 8 + 16 * half + j];
            }
            // B[k,n] = V[kc + 16*half + k, nb + n]; lanes are columns, coalesced per k.
            const float* vcol = Vb + (size_t)(kc + 16 * half) * D_ + nb + m;
            #pragma unroll
            for (int j = 0; j < 16; ++j)
                b[j] = (_Float16)vcol[(size_t)j * D_];
            c = __builtin_amdgcn_wmma_f32_16x16x32_f16(false, a, false, b, (short)0, c, false, false);
        }
        #pragma unroll
        for (int r = 0; r < 8; ++r) {
            const int row = r + 8 * half;
            out[((size_t)bh * S_ + qbase + row) * D_ + nb + m] = c[r] * sInv[row];
        }
    }
}

extern "C" void kernel_launch(void* const* d_in, const int* in_sizes, int n_in,
                              void* d_out, int out_size, void* d_ws, size_t ws_size,
                              hipStream_t stream) {
    (void)in_sizes; (void)n_in; (void)out_size; (void)d_ws; (void)ws_size;
    const float* Q = (const float*)d_in[0];
    const float* K = (const float*)d_in[1];
    const float* V = (const float*)d_in[2];
    float* out  = (float*)d_out;
    float* attn = out + (size_t)B_ * H_ * S_ * D_;   // tuple order: (out, attn)

    dim3 grid(B_ * H_ * (S_ / QTILE));
    attn_fused_kernel<<<grid, NTHREADS, 0, stream>>>(Q, K, V, out, attn);
}
